// FlowGNN_72653666779396
// MI455X (gfx1250) — compile-verified, matrix operand order checked by reading, not run on hardware
//
#include <hip/hip_runtime.h>
#include <hip/hip_bf16.h>

// ---------------------------------------------------------------------------
// FlowGNN for MI455X (gfx1250, wave32).
// Dense GEMMs ([N,128] x [128,{128,64}]) run on v_wmma_f32_16x16x32_bf16.
// Edge scatter: 1 wave / edge, coalesced row loads + global f32 atomics
// (L2-resident: node state is 51MB << 192MB L2).
// ---------------------------------------------------------------------------

typedef __attribute__((ext_vector_type(16))) __bf16 v16bf;
typedef __attribute__((ext_vector_type(8)))  float  v8f;

#define H 128
#define LDSK 136   // padded K stride for LDS weight tile (bank-conflict avoidance)

// ---------------- degree / norm ----------------
__global__ void deg_init_k(float* deg, int N) {
    int t = blockIdx.x * blockDim.x + threadIdx.x;
    if (t < N) deg[t] = 1.0f;                       // self-loop
}
__global__ void deg_acc_k(const int* __restrict__ dst, float* deg, int E) {
    int e = blockIdx.x * blockDim.x + threadIdx.x;
    if (e < E) atomicAdd(&deg[dst[e]], 1.0f);
}
__global__ void deg_rsqrt_k(const float* __restrict__ deg, float* dinv, int N) {
    int t = blockIdx.x * blockDim.x + threadIdx.x;
    if (t < N) dinv[t] = rsqrtf(deg[t]);
}

// ---------------- f32 -> bf16 weight conversion ----------------
__global__ void f2bf_k(const float* __restrict__ src, __bf16* __restrict__ dst, int n) {
    int t = blockIdx.x * blockDim.x + threadIdx.x;
    if (t < n) dst[t] = (__bf16)src[t];
}

// ---------------- input projection: h = x @ Wi + bi ----------------
__global__ void input_proj_k(const float* __restrict__ x, const float* __restrict__ Wi,
                             const float* __restrict__ bi, float* __restrict__ h, int N) {
    size_t t = (size_t)blockIdx.x * blockDim.x + threadIdx.x;
    if (t >= (size_t)N * H) return;
    int n = (int)(t >> 7), f = (int)(t & 127);
    float v = bi[f]
            + x[n * 3 + 0] * Wi[0 * H + f]
            + x[n * 3 + 1] * Wi[1 * H + f]
            + x[n * 3 + 2] * Wi[2 * H + f];
    h[t] = v;
}

// ---------------- WMMA GEMM: Out[N,NCOLS] = A[N,128] * W[128,NCOLS] ----------------
template <int NCOLS, bool RELU_BIAS>
__global__ __launch_bounds__(256) void wmma_gemm_k(const float*  __restrict__ A,
                                                   const __bf16* __restrict__ W,
                                                   const float*  __restrict__ bias,
                                                   float* __restrict__ Out, int nrows) {
    __shared__ __bf16 lds[NCOLS * LDSK];            // transposed: [n][k]
    for (int t = threadIdx.x; t < NCOLS * H; t += 256) {
        int k = t / NCOLS, n = t % NCOLS;
        lds[n * LDSK + k] = W[t];
    }
    __syncthreads();

    const int wave = threadIdx.x >> 5;
    const int lane = threadIdx.x & 31;
    const int half = lane >> 4;                      // ISA lane-layout half
    const int idx  = lane & 15;
    const int rowBase = blockIdx.x * 128 + wave * 16;
    const int arow    = rowBase + idx;
    const int safeRow = (arow < nrows) ? arow : 0;   // keep EXEC all-ones for WMMA
    const float* __restrict__ rowp = A + (size_t)safeRow * H;

    constexpr int NT = NCOLS / 16;
    v8f acc[NT] = {};

#pragma unroll
    for (int ks = 0; ks < 4; ++ks) {                 // K = 128 = 4 * 32
        v16bf a;
        {
            const float* p = rowp + ks * 32 + half * 8;         // K: base+0..7
            const float* q = p + 16;                            // K: base+16..23
#pragma unroll
            for (int j = 0; j < 8; ++j) a[j]     = (__bf16)p[j];
#pragma unroll
            for (int j = 0; j < 8; ++j) a[8 + j] = (__bf16)q[j];
        }
#pragma unroll
        for (int nt = 0; nt < NT; ++nt) {
            v16bf b;
            const __bf16* bp = &lds[(nt * 16 + idx) * LDSK + ks * 32 + half * 8];
            const __bf16* bq = bp + 16;
#pragma unroll
            for (int j = 0; j < 8; ++j) b[j]     = bp[j];
#pragma unroll
            for (int j = 0; j < 8; ++j) b[8 + j] = bq[j];
            acc[nt] = __builtin_amdgcn_wmma_f32_16x16x32_bf16(
                false, a, false, b, (short)0, acc[nt], false, false);
        }
    }

#pragma unroll
    for (int nt = 0; nt < NT; ++nt) {
#pragma unroll
        for (int r = 0; r < 8; ++r) {
            int gr = rowBase + half * 8 + r;         // C layout: VGPR r -> M = r (+8 for hi half)
            if (gr < nrows) {
                int col = nt * 16 + idx;
                float v = acc[nt][r];
                if constexpr (RELU_BIAS) v = fmaxf(v + bias[col], 0.0f);
                Out[(size_t)gr * NCOLS + col] = v;
            }
        }
    }
}

// ---------------- agg = gcn_b + m * dinv^2  (self-loop term) ; zero BN stats ----------------
__global__ void agg_init_k(const float* __restrict__ m, const float* __restrict__ dinv,
                           const float* __restrict__ gb, float* __restrict__ agg,
                           float* __restrict__ stats, int N) {
    size_t t = (size_t)blockIdx.x * blockDim.x + threadIdx.x;
    if (t < 256) stats[t] = 0.0f;
    if (t >= (size_t)N * H) return;
    int n = (int)(t >> 7), f = (int)(t & 127);
    float di = dinv[n];
    agg[t] = gb[f] + m[t] * di * di;
}

// ---------------- edge scatter: agg[dst] += m[src] * dinv[src]*dinv[dst] ----------------
__global__ __launch_bounds__(256) void scatter_k(const float* __restrict__ m,
                                                 const int* __restrict__ src,
                                                 const int* __restrict__ dst,
                                                 const float* __restrict__ dinv,
                                                 float* __restrict__ agg, int E) {
    int gid  = blockIdx.x * blockDim.x + threadIdx.x;
    int e    = gid >> 5;                             // one wave32 per edge
    int lane = gid & 31;
    if (e >= E) return;
    int s = src[e], d = dst[e];
    float norm = dinv[s] * dinv[d];
    const float* mr = m + (size_t)s * H;
    float* ar = agg + (size_t)d * H;
#pragma unroll
    for (int j = 0; j < 4; ++j) {
        int f = lane + j * 32;                       // coalesced across the wave
        atomicAdd(&ar[f], mr[f] * norm);
    }
}

// ---------------- residual + per-feature sum/sumsq (BN stats) ----------------
__global__ __launch_bounds__(256) void residual_stats_k(float* __restrict__ h,
                                                        const float* __restrict__ agg,
                                                        float* __restrict__ stats, int N) {
    __shared__ float ls[256], ls2[256];
    int f   = threadIdx.x & 127;
    int sub = threadIdx.x >> 7;                      // 0 or 1
    int rowBase = blockIdx.x * 128;
    float s = 0.0f, s2 = 0.0f;
    for (int r = sub; r < 128; r += 2) {
        int n = rowBase + r;
        if (n < N) {
            size_t i = (size_t)n * H + f;
            float v = h[i] + agg[i];
            h[i] = v;
            s += v; s2 += v * v;
        }
    }
    ls[threadIdx.x] = s; ls2[threadIdx.x] = s2;
    __syncthreads();
    if (sub == 0) {
        s  += ls[128 + f];
        s2 += ls2[128 + f];
        atomicAdd(&stats[f], s);
        atomicAdd(&stats[128 + f], s2);
    }
}

// ---------------- stats -> (scale, shift) ----------------
__global__ void finalize_stats_k(float* stats, const float* __restrict__ g,
                                 const float* __restrict__ b, float invN) {
    int f = threadIdx.x;                             // 128 threads
    float mu  = stats[f] * invN;
    float var = stats[128 + f] * invN - mu * mu;
    float sc  = g[f] * rsqrtf(var + 1e-5f);
    stats[f]       = sc;
    stats[128 + f] = b[f] - mu * sc;
}

// ---------------- h = relu(h*scale + shift) ----------------
__global__ void bn_relu_k(float* __restrict__ h, const float* __restrict__ stats, size_t total) {
    size_t t = (size_t)blockIdx.x * blockDim.x + threadIdx.x;
    if (t >= total) return;
    int f = (int)(t & 127);
    h[t] = fmaxf(h[t] * stats[f] + stats[128 + f], 0.0f);
}

// ---------------- head: out = h2[N,64] @ W3[64,8] + b3 ----------------
__global__ void head_k(const float* __restrict__ h2, const float* __restrict__ W3,
                       const float* __restrict__ b3, float* __restrict__ out, int N) {
    int t = blockIdx.x * blockDim.x + threadIdx.x;
    if (t >= N * 8) return;
    int n = t >> 3, o = t & 7;
    const float* hr = h2 + (size_t)n * 64;
    float acc = b3[o];
#pragma unroll 8
    for (int k = 0; k < 64; ++k) acc += hr[k] * W3[k * 8 + o];
    out[t] = acc;
}

static inline int cdiv(long long a, int b) { return (int)((a + b - 1) / b); }

extern "C" void kernel_launch(void* const* d_in, const int* in_sizes, int n_in,
                              void* d_out, int out_size, void* d_ws, size_t ws_size,
                              hipStream_t stream) {
    const float* x     = (const float*)d_in[0];
    const int*   eidx  = (const int*)  d_in[1];
    const float* Wi    = (const float*)d_in[2];
    const float* bi    = (const float*)d_in[3];
    const float* gcnW  = (const float*)d_in[4];
    const float* gcnB  = (const float*)d_in[5];
    const float* bnG   = (const float*)d_in[6];
    const float* bnB   = (const float*)d_in[7];
    const float* W1    = (const float*)d_in[8];
    const float* b1    = (const float*)d_in[9];
    const float* W2    = (const float*)d_in[10];
    const float* b2    = (const float*)d_in[11];
    const float* W3    = (const float*)d_in[12];
    const float* b3    = (const float*)d_in[13];
    float* out = (float*)d_out;

    const int N = in_sizes[0] / 3;
    const int E = in_sizes[1] / 2;
    const int* esrc = eidx;
    const int* edst = eidx + E;

    // ---- workspace carve-up ----
    float* h     = (float*)d_ws;                     // N*128
    float* m     = h   + (size_t)N * H;              // N*128
    float* agg   = m   + (size_t)N * H;              // N*128
    float* deg   = agg + (size_t)N * H;              // N
    float* dinv  = deg + N;                          // N
    float* stats = dinv + N;                         // 256
    __bf16* Wb   = (__bf16*)(stats + 256);           // 4*16384 + 16384 + 8192 bf16
    __bf16* W1b  = Wb + 4 * H * H;
    __bf16* W2b  = W1b + H * H;
    float* h1 = agg;                                 // alias (after GCN loop)
    float* h2 = m;                                   // alias

    const size_t NH = (size_t)N * H;

    // degree / normalization constants
    deg_init_k <<<cdiv(N, 256), 256, 0, stream>>>(deg, N);
    deg_acc_k  <<<cdiv(E, 256), 256, 0, stream>>>(edst, deg, E);
    deg_rsqrt_k<<<cdiv(N, 256), 256, 0, stream>>>(deg, dinv, N);

    // weights -> bf16 once per launch
    f2bf_k<<<cdiv(4 * H * H, 256), 256, 0, stream>>>(gcnW, Wb, 4 * H * H);
    f2bf_k<<<cdiv(H * H,     256), 256, 0, stream>>>(W1, W1b, H * H);
    f2bf_k<<<cdiv(H * H / 2, 256), 256, 0, stream>>>(W2, W2b, H * H / 2);

    // input projection
    input_proj_k<<<cdiv((long long)NH, 256), 256, 0, stream>>>(x, Wi, bi, h, N);

    const int gemmBlocks = cdiv(N, 128);
    const float invN = 1.0f / (float)N;

    for (int i = 0; i < 4; ++i) {
        wmma_gemm_k<128, false><<<gemmBlocks, 256, 0, stream>>>(h, Wb + (size_t)i * H * H,
                                                                nullptr, m, N);
        agg_init_k<<<cdiv((long long)NH, 256), 256, 0, stream>>>(m, dinv, gcnB + i * H,
                                                                 agg, stats, N);
        scatter_k<<<cdiv((long long)E * 32, 256), 256, 0, stream>>>(m, esrc, edst, dinv, agg, E);
        residual_stats_k<<<gemmBlocks, 256, 0, stream>>>(h, agg, stats, N);
        finalize_stats_k<<<1, 128, 0, stream>>>(stats, bnG + i * H, bnB + i * H, invN);
        bn_relu_k<<<cdiv((long long)NH, 256), 256, 0, stream>>>(h, stats, NH);
    }

    // MLP head
    wmma_gemm_k<128, true><<<gemmBlocks, 256, 0, stream>>>(h,  W1b, b1, h1, N);
    wmma_gemm_k<64,  true><<<gemmBlocks, 256, 0, stream>>>(h1, W2b, b2, h2, N);
    head_k<<<cdiv((long long)N * 8, 256), 256, 0, stream>>>(h2, W3, b3, out, N);
}
